// Attention_29661044146348
// MI455X (gfx1250) — matrix operand order repaired and checked by reading, not compiled, not run on hardware
//
#include <hip/hip_runtime.h>
#include <hip/hip_bf16.h>

// ---------------------------------------------------------------------------
// Attention block for MI455X (gfx1250, wave32, WMMA bf16 path)
//   GroupNorm -> QKV 1x1 (GEMM) -> streaming-softmax attention -> out proj
// All matmuls use V_WMMA_F32_16X16X32_BF16 (16x16 tile per wave, K-step 32).
// ---------------------------------------------------------------------------

typedef __bf16 bf16x16 __attribute__((ext_vector_type(16)));
typedef float v8f      __attribute__((ext_vector_type(8)));

#define BATCH 8
#define CH    512            // channels (attention head dim)
#define SEQ   4096           // 64*64 spatial
#define OCH   1536           // 3*CH (q,k,v)
#define ATTN_SCALE 0.044194173824159216f  // 512^-0.5

struct Frag {
    union { unsigned int u[8]; bf16x16 v; };
};

// float -> bf16 bits, round-to-nearest-even
__device__ __forceinline__ unsigned short f2bf(float f) {
    unsigned int u = __float_as_uint(f);
    u = (u + 0x7FFFu + ((u >> 16) & 1u)) >> 16;
    return (unsigned short)u;
}

// A-matrix 16x32 bf16 fragment (ISA 7.12.2 layout), row stride ld elems.
// lanes 0-15: M=lane, v in 0..3 -> K=2v,2v+1 ; v in 4..7 -> K=16+2(v-4)+h
// lanes 16-31: same rows, K offset +8 (low group) / +24 (high group)
__device__ __forceinline__ void load_a(Frag& f, const unsigned short* src,
                                       int ld, int lane, int k0) {
    const int m  = lane & 15;
    const int kb = (lane >> 4) * 8;
    const unsigned short* row = src + (size_t)m * ld;
#pragma unroll
    for (int v = 0; v < 8; ++v) {
        int k = k0 + ((v >> 2) * 16) + kb + ((v & 3) * 2);
        f.u[v] = *(const unsigned int*)(row + k);
    }
}

// B-matrix 32x16 bf16 fragment. Element (K,N) at src[n*ld + k] (K contiguous).
// lanes 0-15: N=lane, same K order as load_a; lanes 16-31: K offset +8 / +24.
__device__ __forceinline__ void load_b(Frag& f, const unsigned short* src,
                                       int ld, int lane, int k0) {
    const int n  = lane & 15;
    const int kb = (lane >> 4) * 8;
    const unsigned short* col = src + (size_t)n * ld;
#pragma unroll
    for (int v = 0; v < 8; ++v) {
        int k = k0 + ((v >> 2) * 16) + kb + ((v & 3) * 2);
        f.u[v] = *(const unsigned int*)(col + k);
    }
}

__device__ __forceinline__ v8f wmma_bf16(const Frag& a, const Frag& b, v8f c) {
    return __builtin_amdgcn_wmma_f32_16x16x32_bf16(
        /*neg_a=*/false, a.v, /*neg_b=*/false, b.v,
        /*c_mod=*/(short)0, c, /*reuse_a=*/false, /*reuse_b=*/false);
}

// ---------------------------------------------------------------------------
// Kernel 0: fp32 -> bf16 weight conversion
// ---------------------------------------------------------------------------
__global__ void cvt_bf16_kernel(const float* __restrict__ in,
                                unsigned short* __restrict__ out, int n) {
    int i = blockIdx.x * blockDim.x + threadIdx.x;
    if (i < n) out[i] = f2bf(in[i]);
}

// ---------------------------------------------------------------------------
// Kernel 1: GroupNorm(32).  One block per (batch, group): 16 ch x 4096 spatial.
// Writes normalized activations seq-major bf16:  xn[b][s][c]
// ---------------------------------------------------------------------------
__global__ void groupnorm_kernel(const float* __restrict__ x,
                                 const float* __restrict__ gw,
                                 const float* __restrict__ gb,
                                 unsigned short* __restrict__ xn) {
    const int b  = blockIdx.y;
    const int c0 = blockIdx.x * 16;
    const size_t base = ((size_t)b * CH + c0) * SEQ;   // 16*4096 contiguous f32
    const int tid = threadIdx.x;

    float s1 = 0.f, s2 = 0.f;
    for (int e = tid; e < 16 * SEQ; e += 256) {
        float v = x[base + e];
        s1 += v; s2 += v * v;
    }
    __shared__ float r1[256], r2[256];
    r1[tid] = s1; r2[tid] = s2;
    __syncthreads();
    for (int off = 128; off > 0; off >>= 1) {
        if (tid < off) { r1[tid] += r1[tid + off]; r2[tid] += r2[tid + off]; }
        __syncthreads();
    }
    const float inv_n = 1.0f / (16.0f * SEQ);
    const float mean  = r1[0] * inv_n;
    const float var   = r2[0] * inv_n - mean * mean;
    const float rstd  = rsqrtf(var + 1e-5f);

    for (int e = tid; e < 16 * SEQ; e += 256) {
        int ch = e >> 12;                 // SEQ == 4096
        int s  = e & (SEQ - 1);
        int c  = c0 + ch;
        float y = (x[base + e] - mean) * rstd * gw[c] + gb[c];
        xn[((size_t)b * SEQ + s) * CH + c] = f2bf(y);
    }
}

// ---------------------------------------------------------------------------
// qkv epilogue: route a 16x16 D-tile to q/k ([b][s][c]) or v ([b][c][s])
// ---------------------------------------------------------------------------
__device__ __forceinline__ void store_qkv(int o0, const v8f& acc, int b, int s0,
                                          int n, int half,
                                          const float* __restrict__ bias,
                                          unsigned short* __restrict__ q,
                                          unsigned short* __restrict__ k,
                                          unsigned short* __restrict__ v) {
    const float bs = bias[o0 + n];
    if (o0 < CH) {                                        // -> q [b][s][c]
        unsigned short* dst = q + ((size_t)b * SEQ + s0) * CH + o0;
#pragma unroll
        for (int r = 0; r < 8; ++r) {
            int m = r + 8 * half;
            dst[(size_t)m * CH + n] = f2bf(acc[r] + bs);
        }
    } else if (o0 < 2 * CH) {                             // -> k [b][s][c]
        unsigned short* dst = k + ((size_t)b * SEQ + s0) * CH + (o0 - CH);
#pragma unroll
        for (int r = 0; r < 8; ++r) {
            int m = r + 8 * half;
            dst[(size_t)m * CH + n] = f2bf(acc[r] + bs);
        }
    } else {                                              // -> v [b][c][s]
        unsigned short* dst = v + ((size_t)b * CH + (o0 - 2 * CH) + n) * SEQ + s0;
#pragma unroll
        for (int r = 0; r < 8; ++r) {
            int m = r + 8 * half;
            dst[m] = f2bf(acc[r] + bs);
        }
    }
}

// ---------------------------------------------------------------------------
// Kernel 2: QKV projection.  qkv[b][s][o] = sum_c xn[b][s][c] * W[o][c] + bias
// Each wave computes a 16x32 output strip (two N-tiles) so every A fragment
// feeds two WMMAs.  K=512 -> 16 K-steps -> 32 WMMAs per wave.
// ---------------------------------------------------------------------------
__global__ void qkv_gemm_kernel(const unsigned short* __restrict__ xn,
                                const unsigned short* __restrict__ wq,
                                const float* __restrict__ bias,
                                unsigned short* __restrict__ q,
                                unsigned short* __restrict__ k,
                                unsigned short* __restrict__ v) {
    const int b    = blockIdx.z;
    const int s0   = blockIdx.x * 16;
    const int wave = threadIdx.x >> 5;
    const int lane = threadIdx.x & 31;
    const int o0   = (blockIdx.y * 16 + wave * 2) * 16;   // two adjacent N-tiles

    const unsigned short* A   = xn + ((size_t)b * SEQ + s0) * CH;
    const unsigned short* Bp0 = wq + (size_t)o0 * CH;
    const unsigned short* Bp1 = Bp0 + (size_t)16 * CH;

    v8f acc0 = {}, acc1 = {};
    Frag fa, fb0, fb1;
#pragma unroll 4
    for (int kk = 0; kk < CH; kk += 32) {
        load_a(fa, A, CH, lane, kk);
        load_b(fb0, Bp0, CH, lane, kk);
        load_b(fb1, Bp1, CH, lane, kk);
        acc0 = wmma_bf16(fa, fb0, acc0);
        acc1 = wmma_bf16(fa, fb1, acc1);
    }

    const int n = lane & 15, half = lane >> 4;
    store_qkv(o0,      acc0, b, s0, n, half, bias, q, k, v);
    store_qkv(o0 + 16, acc1, b, s0, n, half, bias, q, k, v);
}

// ---------------------------------------------------------------------------
// Kernel 3: streaming-softmax attention (never materializes the 4096x4096 sim).
// Block = one 16-row query tile, 8 waves (256 thr).
// Q tile (16x512) lives in REGISTERS: 16 A-fragments = 128 VGPRs per wave,
// loaded once, reused by all 32 key-steps (no LDS traffic for Q).
// Per 128-key step:
//   stage 1: wave w computes S(16x16) for its 16 keys (16 WMMAs),
//            exp -> P to LDS, accumulates per-row denominator.
//   stage 2: wave w accumulates num += P(16x128) @ V(128 x its 64-col slice)
//            (4 coltiles x 4 K-steps = 16 WMMAs, P fragments from LDS).
// Epilogue: shfl+LDS denominator reduction, O = num/den -> bf16 [b][s][c].
// ---------------------------------------------------------------------------
__global__ void attn_kernel(const unsigned short* __restrict__ q,
                            const unsigned short* __restrict__ k,
                            const unsigned short* __restrict__ v,
                            unsigned short* __restrict__ o) {
    const int b    = blockIdx.y;
    const int i0   = blockIdx.x * 16;
    const int tid  = threadIdx.x;
    const int wave = tid >> 5;
    const int lane = tid & 31;
    const int n    = lane & 15, half = lane >> 4;

    __shared__ unsigned short Ps[16 * 128];    // 4 KB : exp(S) for 128 keys
    __shared__ float denP[8 * 16];
    __shared__ float denT[16];

    // Q tile -> registers (16 K-step fragments, reused for every key tile)
    Frag qf[16];
    {
        const unsigned short* Aq = q + ((size_t)b * SEQ + i0) * CH;
#pragma unroll
        for (int i = 0; i < 16; ++i) load_a(qf[i], Aq, CH, lane, i * 32);
    }

    v8f acc0 = {}, acc1 = {}, acc2 = {}, acc3 = {};
    float den[8];
#pragma unroll
    for (int r = 0; r < 8; ++r) den[r] = 0.f;

    const int c0w = wave * 64;                             // this wave's V/O cols
    const unsigned short* kb_ = k + (size_t)b * SEQ * CH;
    const unsigned short* vb_ = v + ((size_t)b * CH + c0w) * SEQ;

    for (int jt = 0; jt < SEQ; jt += 128) {
        // ---- stage 1: S = Q @ K^T for this wave's 16 keys ----
        const int j0 = jt + wave * 16;
        const unsigned short* Bk = kb_ + (size_t)j0 * CH;
        // prefetch next K tile / V stripe (global_prefetch_b8)
        if (jt + 128 < SEQ) {
            __builtin_prefetch(Bk + (size_t)128 * CH + lane * 64, 0, 0);
            __builtin_prefetch(Bk + (size_t)128 * CH + 4096 + lane * 64, 0, 0);
            __builtin_prefetch(vb_ + (size_t)(half * 16) * SEQ + jt + 128 + n * 8, 0, 0);
        }
        v8f sacc = {};
        Frag fb;
#pragma unroll
        for (int i = 0; i < 16; ++i) {
            load_b(fb, Bk, CH, lane, i * 32);
            sacc = wmma_bf16(qf[i], fb, sacc);
        }
#pragma unroll
        for (int r = 0; r < 8; ++r) {
            float p = __expf(sacc[r] * ATTN_SCALE);
            den[r] += p;
            int m = r + 8 * half;
            Ps[m * 128 + wave * 16 + n] = f2bf(p);
        }
        __syncthreads();

        // ---- stage 2: num += P @ V (this wave's 64-col slice) ----
#pragma unroll
        for (int ks = 0; ks < 4; ++ks) {
            Frag pa, pb;
            load_a(pa, Ps, 128, lane, ks * 32);
            load_b(pb, vb_ + (size_t)0 * 16 * SEQ + jt, SEQ, lane, ks * 32);
            acc0 = wmma_bf16(pa, pb, acc0);
            load_b(pb, vb_ + (size_t)1 * 16 * SEQ + jt, SEQ, lane, ks * 32);
            acc1 = wmma_bf16(pa, pb, acc1);
            load_b(pb, vb_ + (size_t)2 * 16 * SEQ + jt, SEQ, lane, ks * 32);
            acc2 = wmma_bf16(pa, pb, acc2);
            load_b(pb, vb_ + (size_t)3 * 16 * SEQ + jt, SEQ, lane, ks * 32);
            acc3 = wmma_bf16(pa, pb, acc3);
        }
        __syncthreads();
    }

    // ---- denominator: reduce across the 16 lanes holding each row ----
#pragma unroll
    for (int m = 1; m <= 8; m <<= 1) {
#pragma unroll
        for (int r = 0; r < 8; ++r) den[r] += __shfl_xor(den[r], m, 32);
    }
    if (lane == 0) {
#pragma unroll
        for (int r = 0; r < 8; ++r) denP[wave * 16 + r] = den[r];
    }
    if (lane == 16) {
#pragma unroll
        for (int r = 0; r < 8; ++r) denP[wave * 16 + 8 + r] = den[r];
    }
    __syncthreads();
    if (tid < 16) {
        float s = 0.f;
#pragma unroll
        for (int w = 0; w < 8; ++w) s += denP[w * 16 + tid];
        denT[tid] = s;
    }
    __syncthreads();

    float rd[8];
#pragma unroll
    for (int r = 0; r < 8; ++r) rd[r] = 1.0f / denT[r + 8 * half];

    unsigned short* ob = o + ((size_t)b * SEQ + i0) * CH + c0w;
#pragma unroll
    for (int r = 0; r < 8; ++r) {
        int m = r + 8 * half;
        size_t row = (size_t)m * CH + n;
        ob[row + 0 * 16] = f2bf(acc0[r] * rd[r]);
        ob[row + 1 * 16] = f2bf(acc1[r] * rd[r]);
        ob[row + 2 * 16] = f2bf(acc2[r] * rd[r]);
        ob[row + 3 * 16] = f2bf(acc3[r] * rd[r]);
    }
}

// ---------------------------------------------------------------------------
// Kernel 4: output projection + bias + residual.
// out[b][o][s] = sum_c O[b][s][c]*W2[o][c] + bias[o] + x[b][o][s]
// Each wave computes a 16x32 strip (A fragments feed two WMMAs).
// Epilogue writes contiguous float4 pairs per lane.
// ---------------------------------------------------------------------------
__global__ void proj_kernel(const unsigned short* __restrict__ o_,
                            const unsigned short* __restrict__ w2,
                            const float* __restrict__ bias,
                            const float* __restrict__ x,
                            float* __restrict__ out) {
    const int b    = blockIdx.z;
    const int s0   = blockIdx.x * 16;
    const int wave = threadIdx.x >> 5;
    const int lane = threadIdx.x & 31;
    const int o0   = (blockIdx.y * 16 + wave * 2) * 16;

    const unsigned short* A   = o_ + ((size_t)b * SEQ + s0) * CH;
    const unsigned short* Bp0 = w2 + (size_t)o0 * CH;
    const unsigned short* Bp1 = Bp0 + (size_t)16 * CH;

    v8f acc0 = {}, acc1 = {};
    Frag fa, fb0, fb1;
#pragma unroll 4
    for (int kk = 0; kk < CH; kk += 32) {
        load_a(fa, A, CH, lane, kk);
        load_b(fb0, Bp0, CH, lane, kk);
        load_b(fb1, Bp1, CH, lane, kk);
        acc0 = wmma_bf16(fa, fb0, acc0);
        acc1 = wmma_bf16(fa, fb1, acc1);
    }

    const int n = lane & 15, half = lane >> 4;
#pragma unroll
    for (int t = 0; t < 2; ++t) {
        const v8f& acc = t ? acc1 : acc0;
        const int oo = o0 + t * 16;
        const float bs = bias[oo + n];
        const size_t base = ((size_t)b * CH + oo + n) * SEQ + s0 + 8 * half;
        const float4 xa = *(const float4*)(x + base);
        const float4 xb = *(const float4*)(x + base + 4);
        float4 ra, rb;
        ra.x = acc[0] + bs + xa.x;  ra.y = acc[1] + bs + xa.y;
        ra.z = acc[2] + bs + xa.z;  ra.w = acc[3] + bs + xa.w;
        rb.x = acc[4] + bs + xb.x;  rb.y = acc[5] + bs + xb.y;
        rb.z = acc[6] + bs + xb.z;  rb.w = acc[7] + bs + xb.w;
        *(float4*)(out + base)     = ra;
        *(float4*)(out + base + 4) = rb;
    }
}

// ---------------------------------------------------------------------------
extern "C" void kernel_launch(void* const* d_in, const int* in_sizes, int n_in,
                              void* d_out, int out_size, void* d_ws, size_t ws_size,
                              hipStream_t stream) {
    const float* x     = (const float*)d_in[0];
    const float* gn_w  = (const float*)d_in[1];
    const float* gn_b  = (const float*)d_in[2];
    const float* qkv_w = (const float*)d_in[3];
    const float* qkv_b = (const float*)d_in[4];
    const float* out_w = (const float*)d_in[5];
    const float* out_b = (const float*)d_in[6];
    float* out = (float*)d_out;

    // workspace carve-up (bf16 intermediates)
    const size_t nsc = (size_t)BATCH * SEQ * CH;     // 16M elems
    char* ws = (char*)d_ws;
    unsigned short* xn = (unsigned short*)ws;  ws += nsc * 2;
    unsigned short* qb = (unsigned short*)ws;  ws += nsc * 2;
    unsigned short* kb = (unsigned short*)ws;  ws += nsc * 2;
    unsigned short* vb = (unsigned short*)ws;  ws += nsc * 2;
    unsigned short* ob = (unsigned short*)ws;  ws += nsc * 2;
    unsigned short* wq = (unsigned short*)ws;  ws += (size_t)OCH * CH * 2;
    unsigned short* w2 = (unsigned short*)ws;  ws += (size_t)CH * CH * 2;

    cvt_bf16_kernel<<<(OCH * CH + 255) / 256, 256, 0, stream>>>(qkv_w, wq, OCH * CH);
    cvt_bf16_kernel<<<(CH * CH + 255) / 256, 256, 0, stream>>>(out_w, w2, CH * CH);

    groupnorm_kernel<<<dim3(32, BATCH), 256, 0, stream>>>(x, gn_w, gn_b, xn);

    qkv_gemm_kernel<<<dim3(SEQ / 16, OCH / 256, BATCH), 256, 0, stream>>>(
        xn, wq, qkv_b, qb, kb, vb);

    attn_kernel<<<dim3(SEQ / 16, BATCH), 256, 0, stream>>>(qb, kb, vb, ob);

    proj_kernel<<<dim3(SEQ / 16, CH / 256, BATCH), 256, 0, stream>>>(
        ob, w2, out_b, x, out);
}
